// ATTenModel_38809324486671
// MI455X (gfx1250) — matrix operand order repaired and checked by reading, not compiled, not run on hardware
//
#include <hip/hip_runtime.h>
#include <stdint.h>

// ---------------------------------------------------------------------------
// MI455X (gfx1250) fused MLP + entmax15 straight-through kernel.
// Memory-bound problem: 419 MB of x @ 23.3 TB/s ~= 18 us floor. We keep the
// GEMM under that roof with bf16x3 split-precision WMMA (f32 accumulate) and
// full layer fusion. Weights are staged to LDS once per workgroup with the
// Tensor Data Mover (tensor_load_to_lds) and reused by all 8 waves.
// ---------------------------------------------------------------------------

typedef __attribute__((ext_vector_type(16))) __bf16   v16bf;
typedef __attribute__((ext_vector_type(8)))  float    v8f;
typedef __attribute__((ext_vector_type(4)))  unsigned u32x4;
typedef __attribute__((ext_vector_type(8)))  int      i32x8;
typedef __attribute__((ext_vector_type(4)))  int      i32x4;

#define D_IN   200
#define D_EMB  100
#define KT     7                         // 7*32 = 224 >= 200 (K tiles)
#define NT     7                         // 7*16 = 112 >= 100 (N tiles)
#define B_DWORDS (2*NT*KT*32*8)          // hi+lo fragments: 25088 dwords = 100352 B
#define ROWS_PER_BLOCK 128               // 8 waves x 16 rows

__device__ __forceinline__ unsigned short f32_to_bf16(float f) {
  unsigned u = __float_as_uint(f);
  u += 0x7fffu + ((u >> 16) & 1u);       // round to nearest even
  return (unsigned short)(u >> 16);
}
__device__ __forceinline__ float bf16_to_f32(unsigned short h) {
  return __uint_as_float(((unsigned)h) << 16);
}

// ---------------------------------------------------------------------------
// Prep: pack w_out (100x200 f32) into WMMA B-fragment order, split into
// bf16 hi/lo planes, zero-padded to 112x224.
// B 32x16 order used here (k contiguous, same as A): lane l -> n = l&15, khalf = l>>4; dword r holds
// K = 16*khalf + 2r, 2r+1 (K ascending within dword, low half first).
// Buffer order: [h(hi/lo)][nt][kt][lane][r] dwords.
// ---------------------------------------------------------------------------
__global__ void prep_wfrag(const float* __restrict__ w_out,
                           unsigned* __restrict__ frag) {
  int idx = blockIdx.x * 256 + threadIdx.x;
  if (idx >= B_DWORDS) return;
  int r    = idx & 7;
  int t    = idx >> 3;
  int lane = t & 31;  t >>= 5;
  int kt   = t % KT;  t /= KT;
  int nt   = t % NT;
  int h    = t / NT;                     // 0 = hi plane, 1 = lo plane
  int n = nt * 16 + (lane & 15);
  int k = kt * 32 + (lane >> 4) * 16 + 2 * r;
  float v0 = (n < D_EMB && k     < D_IN) ? w_out[n * D_IN + k]     : 0.0f;
  float v1 = (n < D_EMB && k + 1 < D_IN) ? w_out[n * D_IN + k + 1] : 0.0f;
  unsigned short h0 = f32_to_bf16(v0);
  unsigned short h1 = f32_to_bf16(v1);
  unsigned short o0, o1;
  if (h == 0) { o0 = h0; o1 = h1; }
  else {
    o0 = f32_to_bf16(v0 - bf16_to_f32(h0));
    o1 = f32_to_bf16(v1 - bf16_to_f32(h1));
  }
  frag[idx] = (unsigned)o0 | ((unsigned)o1 << 16);
}

union Frag8 { unsigned u[8]; v16bf v; };

// ---------------------------------------------------------------------------
// Fused kernel: per 16-row tile per wave:
//   A fragments (bf16 hi/lo) built from x per the 16-bit A 16x32 layout,
//   C[nt] = sum_kt ( Ah*Bh + Ah*Bl + Al*Bh )  via v_wmma_f32_16x16x32_bf16,
//   epilogue folds b_out + relu + w_cat, butterfly-reduce, entmax bisect.
// ---------------------------------------------------------------------------
__global__ __launch_bounds__(256) void fused_mlp_entmax(
    const float* __restrict__ x,
    const unsigned* __restrict__ wfrag,
    const float* __restrict__ b_out,
    const float* __restrict__ w_cat,
    const float* __restrict__ b_cat,
    const float* __restrict__ w2,
    const float* __restrict__ b2,
    float* __restrict__ out)
{
  extern __shared__ unsigned smem[];
  unsigned* sB   = smem;                        // B_DWORDS dwords (100 KB)
  float*   sBias = (float*)(smem + B_DWORDS);   // 112 (b_out padded)
  float*   sW0   = sBias + 112;                 // 112 (w_cat row 0 padded)
  float*   sW1   = sW0 + 112;                   // 112 (w_cat row 1 padded)

  const int tid = threadIdx.x;

  if (tid < 112) {
    sBias[tid] = (tid < D_EMB) ? b_out[tid]         : 0.0f;
    sW0[tid]   = (tid < D_EMB) ? w_cat[tid]         : 0.0f;
    sW1[tid]   = (tid < D_EMB) ? w_cat[D_EMB + tid] : 0.0f;
  }

  // ---- TDM: DMA the 100 KB fragment buffer into LDS (1-D tile) ----
  if (tid == 0) {
    unsigned lds_off = (unsigned)(size_t)(void*)sB;  // LDS byte offset (addr[31:0])
    unsigned long long ga = (unsigned long long)(size_t)wfrag;
    // D# group 0: count=1 | lds_addr | global_addr | type=2
    u32x4 g0 = { 1u,
                 lds_off,
                 (unsigned)(ga & 0xffffffffull),
                 (unsigned)((ga >> 32) & 0x1ffffffull) | (2u << 30) };
    // D# group 1: data_size=4B; tensor_dim0=tile_dim0=B_DWORDS; dim1=1
    i32x8 g1 = { (int)(2u << 16),                               // data_size=2 (4B)
                 (int)(((unsigned)B_DWORDS & 0xffffu) << 16),   // tensor_dim0[15:0]
                 (int)(((unsigned)B_DWORDS >> 16) | (1u << 16)),// dim0[31:16] | tensor_dim1=1
                 (int)(((unsigned)B_DWORDS & 0xffffu) << 16),   // tile_dim0
                 0,                                             // tile_dim1=0, tile_dim2=0
                 (int)(unsigned)B_DWORDS,                       // tensor_dim0_stride lo32
                 0, 0 };
    i32x4 gz4 = { 0, 0, 0, 0 };
    i32x8 gz8 = { 0, 0, 0, 0, 0, 0, 0, 0 };
    // amdgpu-toolchain (clang-23) 6-arg form: (g0, g1, g2, g3, g4, cpol)
    __builtin_amdgcn_tensor_load_to_lds(g0, g1, gz4, gz4, gz8, 0);
    __builtin_amdgcn_s_wait_tensorcnt(0);
  }
  __syncthreads();

  const int lane = tid & 31;
  const int wave = tid >> 5;
  const int mrow = lane & 15;              // row within 16-tile / C column
  const int kh   = lane >> 4;
  const size_t row = (size_t)blockIdx.x * ROWS_PER_BLOCK + wave * 16 + mrow;
  const float2* xr = (const float2*)(x + row * D_IN);

  // ---- Build A fragments (16-bit A 16x32 layout), hi/lo bf16 split ----
  Frag8 Ah[KT], Al[KT];
  #pragma unroll
  for (int kt = 0; kt < KT; ++kt) {
    #pragma unroll
    for (int r = 0; r < 8; ++r) {
      int k = kt * 32 + 16 * kh + 2 * r;
      float a0 = 0.0f, a1 = 0.0f;
      if (k < D_IN) { float2 p = xr[k >> 1]; a0 = p.x; a1 = p.y; }
      unsigned short h0 = f32_to_bf16(a0), h1 = f32_to_bf16(a1);
      Ah[kt].u[r] = (unsigned)h0 | ((unsigned)h1 << 16);
      unsigned short l0 = f32_to_bf16(a0 - bf16_to_f32(h0));
      unsigned short l1 = f32_to_bf16(a1 - bf16_to_f32(h1));
      Al[kt].u[r] = (unsigned)l0 | ((unsigned)l1 << 16);
    }
  }

  float acc0[8], acc1[8];
  #pragma unroll
  for (int v = 0; v < 8; ++v) { acc0[v] = 0.0f; acc1[v] = 0.0f; }

  for (int nt = 0; nt < NT; ++nt) {
    v8f c = {};
    #pragma unroll
    for (int kt = 0; kt < KT; ++kt) {
      Frag8 Bh, Bl;
      const uint4* ph = (const uint4*)(sB + (nt * KT + kt) * 256 + lane * 8);
      const uint4* pl = (const uint4*)(sB + (NT * KT + nt * KT + kt) * 256 + lane * 8);
      *(uint4*)&Bh.u[0] = ph[0];  *(uint4*)&Bh.u[4] = ph[1];
      *(uint4*)&Bl.u[0] = pl[0];  *(uint4*)&Bl.u[4] = pl[1];
      c = __builtin_amdgcn_wmma_f32_16x16x32_bf16(false, Ah[kt].v, false, Bh.v,
                                                  (short)0, c, false, false);
      c = __builtin_amdgcn_wmma_f32_16x16x32_bf16(false, Ah[kt].v, false, Bl.v,
                                                  (short)0, c, false, false);
      c = __builtin_amdgcn_wmma_f32_16x16x32_bf16(false, Al[kt].v, false, Bh.v,
                                                  (short)0, c, false, false);
    }
    // epilogue: h = relu(c + b_out[j]); acc += h * w_cat[:,j]
    int j = nt * 16 + mrow;                // C col n = lane&15
    float bb = sBias[j], wc0 = sW0[j], wc1 = sW1[j];
    #pragma unroll
    for (int v = 0; v < 8; ++v) {
      float hval = fmaxf(c[v] + bb, 0.0f);
      acc0[v] += hval * wc0;
      acc1[v] += hval * wc1;
    }
  }

  // ---- reduce over the 16 n-lanes of each half-wave (C row m = v + 8*half) ----
  #pragma unroll
  for (int v = 0; v < 8; ++v) {
    #pragma unroll
    for (int off = 8; off >= 1; off >>= 1) {
      acc0[v] += __shfl_xor(acc0[v], off, 16);
      acc1[v] += __shfl_xor(acc1[v], off, 16);
    }
  }
  // redistribute: lane i (0..15) takes row i  (v = i&7, half = i>>3)
  int src = ((lane >> 3) & 1) << 4;
  float z0 = 0.0f, z1 = 0.0f;
  #pragma unroll
  for (int v = 0; v < 8; ++v) {
    float t0 = __shfl(acc0[v], src, 32);
    float t1 = __shfl(acc1[v], src, 32);
    if ((lane & 7) == v) { z0 = t0; z1 = t1; }
  }

  if (lane < 16) {
    z0 = __fadd_rn(z0, b_cat[0]);
    z1 = __fadd_rn(z1, b_cat[1]);
    // entmax 1.5 bisection (faithful fp32 replication, no FMA contraction)
    float X0 = __fmul_rn(z0, 0.5f);
    float X1 = __fmul_rn(z1, 0.5f);
    float mx  = fmaxf(X0, X1);
    float tau = __fsub_rn(mx, 1.0f);
    float thi = __fsub_rn(mx, 0.70710678118654752440f);  // (1/2)^0.5
    float dm  = __fsub_rn(thi, tau);
    float r0  = fmaxf(__fsub_rn(X0, tau), 0.0f);
    float r1  = fmaxf(__fsub_rn(X1, tau), 0.0f);
    float flo = __fsub_rn(__fadd_rn(__fmul_rn(r0, r0), __fmul_rn(r1, r1)), 1.0f);
    #pragma unroll 1
    for (int it = 0; it < 50; ++it) {
      dm = __fmul_rn(dm, 0.5f);
      float tm = __fadd_rn(tau, dm);
      float s0 = fmaxf(__fsub_rn(X0, tm), 0.0f);
      float s1 = fmaxf(__fsub_rn(X1, tm), 0.0f);
      float fm = __fsub_rn(__fadd_rn(__fmul_rn(s0, s0), __fmul_rn(s1, s1)), 1.0f);
      if (__fmul_rn(fm, flo) >= 0.0f) tau = tm;
    }
    float tf = __fadd_rn(tau, __fmul_rn(dm, 0.5f));
    float p0 = fmaxf(__fsub_rn(X0, tf), 0.0f); p0 = __fmul_rn(p0, p0);
    float p1 = fmaxf(__fsub_rn(X1, tf), 0.0f); p1 = __fmul_rn(p1, p1);
    float ps = __fadd_rn(p0, p1);
    p0 = __fdiv_rn(p0, ps);
    p1 = __fdiv_rn(p1, ps);
    float y0 = (p0 >= p1) ? 1.0f : 0.0f;     // argmax (first-max, like jnp)
    float y1 = 1.0f - y0;
    float a0 = __fadd_rn(__fsub_rn(y0, p0), p0);   // straight-through residual
    float a1 = __fadd_rn(__fsub_rn(y1, p1), p1);
    float res = __fadd_rn(__fadd_rn(__fmul_rn(a0, w2[0]),
                                    __fmul_rn(a1, w2[1])), b2[0]);
    out[(size_t)blockIdx.x * ROWS_PER_BLOCK + wave * 16 + lane] = res;
  }
}

extern "C" void kernel_launch(void* const* d_in, const int* in_sizes, int n_in,
                              void* d_out, int out_size, void* d_ws, size_t ws_size,
                              hipStream_t stream) {
  (void)n_in; (void)out_size; (void)ws_size;
  const float* x     = (const float*)d_in[0];
  const float* w_out = (const float*)d_in[1];
  const float* b_out = (const float*)d_in[2];
  const float* w_cat = (const float*)d_in[3];
  const float* b_cat = (const float*)d_in[4];
  const float* w2    = (const float*)d_in[5];
  const float* b2    = (const float*)d_in[6];
  unsigned* frag = (unsigned*)d_ws;

  prep_wfrag<<<(B_DWORDS + 255) / 256, 256, 0, stream>>>(w_out, frag);

  const int nrows  = in_sizes[0] / D_IN;           // 524288
  const int blocks = nrows / ROWS_PER_BLOCK;       // 4096
  const size_t shmem = (size_t)B_DWORDS * 4 + 3 * 112 * sizeof(float);
  fused_mlp_entmax<<<blocks, 256, shmem, stream>>>(x, frag, b_out, w_cat, b_cat,
                                                   w2, b2, (float*)d_out);
}
